// Attention_22462678958397
// MI455X (gfx1250) — compile-verified, matrix operand order checked by reading, not run on hardware
//
#include <hip/hip_runtime.h>
#include <hip/hip_bf16.h>
#include <math.h>

typedef __attribute__((ext_vector_type(16))) _Float16 v16h;
typedef __attribute__((ext_vector_type(8)))  _Float16 v8h;
typedef __attribute__((ext_vector_type(8)))  float    v8f;

#define CC 116   // channels / seq-len
#define PD 128   // padded to 8x16 WMMA tiles
#define RK 6     // SVD rank
#define ITERS 32 // orthogonal-iteration steps

// ---------------------------------------------------------------------------
// Fragment loader. src is a row-major [PD x PD] f16 matrix (LDS or global).
// Serves BOTH operands of D = L * R^T:
//   A frag (16x32, row-major L):  rows row0..row0+15, K slab k0..k0+31
//   B frag (32x16) taken from row-major R as B[k][n] = R[n][k]
// Per CDNA5 ISA 16-bit A layout: lane m=lane&15, g=lane>>4;
//   halves 0..7  -> K = k0 + 8g .. +7
//   halves 8..15 -> K = k0 + 16 + 8g .. +7
// Both 8-half chunks are contiguous 16-byte loads.
// ---------------------------------------------------------------------------
__device__ __forceinline__ v16h load_frag(const _Float16* __restrict__ src,
                                          int row0, int k0, int lane)
{
    const int m = lane & 15;
    const int g = lane >> 4;
    const _Float16* p = src + (row0 + m) * PD + k0 + 8 * g;
    v8h lo = *(const v8h*)(p);
    v8h hi = *(const v8h*)(p + 16);
    v16h f;
#pragma unroll
    for (int i = 0; i < 8; ++i) { f[i] = lo[i]; f[8 + i] = hi[i]; }
    return f;
}

// D(16x128 stripe for this wave) = L * R^T, f32 accumulate.
__device__ __forceinline__ void gemm_ABt(const _Float16* __restrict__ L,
                                         const _Float16* __restrict__ R,
                                         int wave, int lane, v8f acc[8])
{
#pragma unroll
    for (int n = 0; n < 8; ++n) { v8f z = {}; acc[n] = z; }
#pragma unroll
    for (int kk = 0; kk < 4; ++kk) {
        const int k0 = kk * 32;
        v16h a = load_frag(L, wave * 16, k0, lane);
#pragma unroll
        for (int n = 0; n < 8; ++n) {
            v16h b = load_frag(R, n * 16, k0, lane);
            acc[n] = __builtin_amdgcn_wmma_f32_16x16x32_f16(
                false, a, false, b, (short)0, acc[n], false, false);
        }
    }
}

// ---------------------------------------------------------------------------
// Pre-kernel: convert weights/biases once into zero-padded buffers in d_ws.
// ---------------------------------------------------------------------------
__global__ void prep_weights(const float* __restrict__ W1, const float* __restrict__ b1,
                             const float* __restrict__ W2, const float* __restrict__ b2,
                             _Float16* __restrict__ W1h, _Float16* __restrict__ W2h,
                             float* __restrict__ b1f, float* __restrict__ b2f)
{
    const int i = blockIdx.x * 256 + threadIdx.x;
    if (i < PD * PD) {
        int r = i >> 7, c = i & 127;
        W1h[i] = (_Float16)((r < CC && c < CC) ? W1[r * CC + c] : 0.0f);
    } else if (i < 2 * PD * PD) {
        int j = i - PD * PD;
        int r = j >> 7, c = j & 127;
        W2h[j] = (_Float16)((r < CC && c < CC) ? W2[r * CC + c] : 0.0f);
    } else if (i < 2 * PD * PD + PD) {
        int c = i - 2 * PD * PD;
        b1f[c] = (c < CC) ? b1[c] : 0.0f;
    } else if (i < 2 * PD * PD + 2 * PD) {
        int c = i - 2 * PD * PD - PD;
        b2f[c] = (c < CC) ? b2[c] : 0.0f;
    }
}

// ---------------------------------------------------------------------------
// Main kernel: one workgroup (8 wave32) per batch element.
// ---------------------------------------------------------------------------
__global__ __launch_bounds__(256)
void attn_svd_kernel(const float* __restrict__ Z, const float* __restrict__ X,
                     const _Float16* __restrict__ W1h, const _Float16* __restrict__ W2h,
                     const float* __restrict__ b1f, const float* __restrict__ b2f,
                     const int* __restrict__ psvd, float* __restrict__ out)
{
    __shared__ _Float16 s_a[PD * PD];   // Xh -> Zh -> Mh
    __shared__ _Float16 s_q[PD * PD];   // Qh -> attn_h -> Ph
    __shared__ _Float16 s_v[PD * PD];   // Vth -> Mt_h
    __shared__ float    s_f[CC * CC];   // S scores -> G
    __shared__ float    s_y[PD * RK];   // subspace block
    __shared__ float    s_y2[PD * RK];
    __shared__ float    s_gram[RK * RK];
    __shared__ float    s_L[RK * RK];

    const int tid  = threadIdx.x;
    const int lane = tid & 31;
    const int wave = tid >> 5;
    const int g    = lane >> 4;
    const int b    = blockIdx.x;
    const size_t base = (size_t)b * CC * CC;
    const bool dosvd = (psvd[0] != 0);

    // ---- X -> Xh (zero-padded f16) ----
    for (int i = tid; i < PD * PD; i += 256) {
        int r = i >> 7, c = i & 127;
        float v = (r < CC && c < CC) ? X[base + r * CC + c] : 0.0f;
        s_a[i] = (_Float16)v;
    }
    __syncthreads();

    // ---- GEMM1: Q = Xh * W1h^T + b1 (masked, f16 -> s_q) ----
    {
        v8f acc[8];
        gemm_ABt(s_a, W1h, wave, lane, acc);
#pragma unroll
        for (int n = 0; n < 8; ++n) {
            const int N = n * 16 + (lane & 15);
            const float bias = b1f[N];
#pragma unroll
            for (int r = 0; r < 8; ++r) {
                const int M = wave * 16 + 8 * g + r;
                float v = (M < CC && N < CC) ? (acc[n][r] + bias) : 0.0f;
                s_q[M * PD + N] = (_Float16)v;
            }
        }
    }
    __syncthreads();

    // ---- Z -> Zh (reuse s_a) ; GEMM2: S = Q * Q^T -> s_f (f32) ----
    for (int i = tid; i < PD * PD; i += 256) {
        int r = i >> 7, c = i & 127;
        float v = (r < CC && c < CC) ? Z[base + r * CC + c] : 0.0f;
        s_a[i] = (_Float16)v;
    }
    {
        v8f acc[8];
        gemm_ABt(s_q, s_q, wave, lane, acc);
#pragma unroll
        for (int n = 0; n < 8; ++n) {
            const int N = n * 16 + (lane & 15);
#pragma unroll
            for (int r = 0; r < 8; ++r) {
                const int M = wave * 16 + 8 * g + r;
                if (M < CC && N < CC) s_f[M * CC + N] = acc[n][r];
            }
        }
    }
    __syncthreads();

    // ---- GEMM3: Vt = Zh * W2h^T + b2 -> s_v ; softmax(S) -> attn_h in s_q ----
    {
        v8f acc[8];
        gemm_ABt(s_a, W2h, wave, lane, acc);
#pragma unroll
        for (int n = 0; n < 8; ++n) {
            const int N = n * 16 + (lane & 15);
            const float bias = b2f[N];
#pragma unroll
            for (int r = 0; r < 8; ++r) {
                const int M = wave * 16 + 8 * g + r;
                float v = (M < CC && N < CC) ? (acc[n][r] + bias) : 0.0f;
                s_v[M * PD + N] = (_Float16)v;
            }
        }
    }
    if (tid < PD) {
        const int row = tid;
        if (row < CC) {
            float mx = -1e30f;
            for (int k = 0; k < CC; ++k) mx = fmaxf(mx, s_f[row * CC + k]);
            float sum = 0.0f;
            for (int k = 0; k < CC; ++k) sum += __expf(s_f[row * CC + k] - mx);
            const float inv = 1.0f / sum;
            for (int k = 0; k < PD; ++k) {
                float p = (k < CC) ? __expf(s_f[row * CC + k] - mx) * inv : 0.0f;
                s_q[row * PD + k] = (_Float16)p;
            }
        } else {
            for (int k = 0; k < PD; ++k) s_q[row * PD + k] = (_Float16)0.0f;
        }
    }
    __syncthreads();

    // ---- GEMM4: Mmat = attn_h * Vt^T ----
    v8f macc[8];
    gemm_ABt(s_q, s_v, wave, lane, macc);

    if (!dosvd) {
        // out[b,l,i] = Mmat[i,l]
#pragma unroll
        for (int n = 0; n < 8; ++n) {
            const int N = n * 16 + (lane & 15);
#pragma unroll
            for (int r = 0; r < 8; ++r) {
                const int M = wave * 16 + 8 * g + r;
                if (M < CC && N < CC) out[base + N * CC + M] = macc[n][r];
            }
        }
        return;
    }

    // masked f16 M -> s_a (row-major)
#pragma unroll
    for (int n = 0; n < 8; ++n) {
        const int N = n * 16 + (lane & 15);
#pragma unroll
        for (int r = 0; r < 8; ++r) {
            const int M = wave * 16 + 8 * g + r;
            float v = (M < CC && N < CC) ? macc[n][r] : 0.0f;
            s_a[M * PD + N] = (_Float16)v;
        }
    }
    __syncthreads();

    // M^T (f16) -> s_v ; GEMM5: G = M * M^T -> s_f (f32)
#pragma unroll
    for (int n = 0; n < 8; ++n) {
        const int N = n * 16 + (lane & 15);
#pragma unroll
        for (int r = 0; r < 8; ++r) {
            const int M = wave * 16 + 8 * g + r;
            float v = (M < CC && N < CC) ? macc[n][r] : 0.0f;
            s_v[N * PD + M] = (_Float16)v;
        }
    }
    {
        v8f acc[8];
        gemm_ABt(s_a, s_a, wave, lane, acc);
#pragma unroll
        for (int n = 0; n < 8; ++n) {
            const int N = n * 16 + (lane & 15);
#pragma unroll
            for (int r = 0; r < 8; ++r) {
                const int M = wave * 16 + 8 * g + r;
                if (M < CC && N < CC) s_f[M * CC + N] = acc[n][r];
            }
        }
    }
    __syncthreads();

    // ---- Orthogonal iteration with deterministic CholQR: top-RK eigvecs of G ----
    for (int i = tid; i < PD * RK; i += 256) {
        int r = i / RK, c = i % RK;
        float v = 0.0f;
        if (r < CC) {
            unsigned h = (unsigned)(r * RK + c) * 2654435761u;
            v = (float)((h >> 8) & 0xFFFF) / 65536.0f - 0.5f;
        }
        s_y[i] = v;
    }
    __syncthreads();

    for (int it = 0; it < ITERS; ++it) {
        // Y2 = G * Y
        for (int i = tid; i < CC * RK; i += 256) {
            int r = i / RK, c = i % RK;
            float s = 0.0f;
            for (int k = 0; k < CC; ++k) s += s_f[r * CC + k] * s_y[k * RK + c];
            s_y2[r * RK + c] = s;
        }
        __syncthreads();
        // Gram = Y2^T Y2 (21 unique pairs, one thread each -> deterministic)
        if (tid < (RK * (RK + 1)) / 2) {
            int t = tid, aa = 0;
            while (t >= RK - aa) { t -= RK - aa; ++aa; }
            const int bb = aa + t;
            float s = 0.0f;
            for (int k = 0; k < CC; ++k) s += s_y2[k * RK + aa] * s_y2[k * RK + bb];
            s_gram[aa * RK + bb] = s;
        }
        __syncthreads();
        // Cholesky (thread 0)
        if (tid == 0) {
            float Lf[RK][RK];
            for (int j = 0; j < RK; ++j) {
                float d = s_gram[j * RK + j];
                for (int k = 0; k < j; ++k) d -= Lf[j][k] * Lf[j][k];
                const float Ljj = sqrtf(fmaxf(d, 1e-12f));
                Lf[j][j] = Ljj;
                const float inv = 1.0f / Ljj;
                for (int i2 = j + 1; i2 < RK; ++i2) {
                    float s = s_gram[j * RK + i2]; // G[i2][j], stored upper
                    for (int k = 0; k < j; ++k) s -= Lf[i2][k] * Lf[j][k];
                    Lf[i2][j] = s * inv;
                }
            }
            for (int i2 = 0; i2 < RK; ++i2)
                for (int j = 0; j <= i2; ++j) s_L[i2 * RK + j] = Lf[i2][j];
        }
        __syncthreads();
        // Y = Y2 * L^{-T}: per-row forward substitution L x = y2_row^T
        for (int r = tid; r < PD; r += 256) {
            if (r < CC) {
                float x[RK];
                for (int i2 = 0; i2 < RK; ++i2) {
                    float s = s_y2[r * RK + i2];
                    for (int j = 0; j < i2; ++j) s -= s_L[i2 * RK + j] * x[j];
                    x[i2] = s / s_L[i2 * RK + i2];
                }
                for (int i2 = 0; i2 < RK; ++i2) s_y[r * RK + i2] = x[i2];
            } else {
                for (int i2 = 0; i2 < RK; ++i2) s_y[r * RK + i2] = 0.0f;
            }
        }
        __syncthreads();
    }

    // ---- P = Y * Y^T (symmetric projector), f16 -> s_q (pad rows of Y are 0) ----
    for (int i = tid; i < PD * PD; i += 256) {
        int r = i >> 7, c = i & 127;
        float s = 0.0f;
        for (int t = 0; t < RK; ++t) s += s_y[r * RK + t] * s_y[c * RK + t];
        s_q[i] = (_Float16)s;
    }
    __syncthreads();

    // ---- GEMM6: Rt = M^T * P = s_v * Ph^T (P symmetric) ; Rt[l][c] -> out[b,l,c] ----
    {
        v8f acc[8];
        gemm_ABt(s_v, s_q, wave, lane, acc);
#pragma unroll
        for (int n = 0; n < 8; ++n) {
            const int N = n * 16 + (lane & 15);
#pragma unroll
            for (int r = 0; r < 8; ++r) {
                const int M = wave * 16 + 8 * g + r;
                if (M < CC && N < CC) out[base + M * CC + N] = acc[n][r];
            }
        }
    }
}

extern "C" void kernel_launch(void* const* d_in, const int* in_sizes, int n_in,
                              void* d_out, int out_size, void* d_ws, size_t ws_size,
                              hipStream_t stream)
{
    (void)in_sizes; (void)n_in; (void)out_size; (void)ws_size;
    const float* Z   = (const float*)d_in[0];
    const float* X   = (const float*)d_in[1];
    const float* W1  = (const float*)d_in[2];
    const float* b1  = (const float*)d_in[3];
    const float* W2  = (const float*)d_in[4];
    const float* b2  = (const float*)d_in[5];
    const int* psvd  = (const int*)d_in[6];

    _Float16* W1h = (_Float16*)d_ws;
    _Float16* W2h = W1h + PD * PD;
    float*    b1f = (float*)(W2h + PD * PD);
    float*    b2f = b1f + PD;

    prep_weights<<<129, 256, 0, stream>>>(W1, b1, W2, b2, W1h, W2h, b1f, b2f);
    attn_svd_kernel<<<2048, 256, 0, stream>>>(Z, X, W1h, W2h, b1f, b2f, psvd,
                                              (float*)d_out);
}
